// RoPEMHA_27075473834413
// MI455X (gfx1250) — matrix operand order repaired and checked by reading, not compiled, not run on hardware
//
#include <hip/hip_runtime.h>
#include <hip/hip_bf16.h>
#include <math.h>

// ---------------- types for WMMA fragments (wave32, gfx1250) ----------------
typedef __attribute__((ext_vector_type(16))) _Float16 v16h;
typedef __attribute__((ext_vector_type(8)))  _Float16 v8h;
typedef __attribute__((ext_vector_type(8)))  float    v8f;

union Frag16 {
    v16h h16;
    v8h  h8[2];
};

__device__ inline v8f wmma_f16(v16h a, v16h b, v8f c) {
    // D = A(16x32 f16) * B(32x16 f16) + C(16x16 f32)
    return __builtin_amdgcn_wmma_f32_16x16x32_f16(
        false, a, false, b, (short)0, c, false, false);
}

__device__ inline float redmax16(float v) {
    // reduce across the 16-lane half-group (xor masks stay inside the half)
    #pragma unroll
    for (int m = 1; m < 16; m <<= 1) v = fmaxf(v, __shfl_xor(v, m, 32));
    return v;
}
__device__ inline float redsum16(float v) {
    #pragma unroll
    for (int m = 1; m < 16; m <<= 1) v += __shfl_xor(v, m, 32);
    return v;
}

// ---------------- problem constants ----------------
#define BB   16
#define MM   2048
#define DD   768
#define HH   12
#define HD   64
#define HALFD 32
#define NM   (BB * MM)          // 32768 rows
#define NMD  ((size_t)NM * DD)  // 25,165,824 elements

// ---------------- conversion kernels ----------------
__global__ __launch_bounds__(256) void k_cvt_f32_to_f16(
    const float* __restrict__ in, _Float16* __restrict__ out, size_t n) {
    size_t i = (size_t)blockIdx.x * 256 + threadIdx.x;
    if (i < n) out[i] = (_Float16)in[i];
}

// W (K x N) f32 -> Wt (N x K) f16  (so B-operand loads are contiguous)
__global__ __launch_bounds__(256) void k_w_transpose_f16(
    const float* __restrict__ W, _Float16* __restrict__ Wt, int dim) {
    size_t i = (size_t)blockIdx.x * 256 + threadIdx.x;
    size_t tot = (size_t)dim * dim;
    if (i >= tot) return;
    int k = (int)(i / dim);
    int n = (int)(i % dim);
    Wt[(size_t)n * dim + k] = (_Float16)W[i];
}

// ---------------- WMMA GEMM:  C(Mr x N) = A(Mr x K) * Bt(N x K)^T + bias ---
// 2x2 register blocking: each wave owns a 32x32 output tile (4 accumulators),
// 4 wmma per K-step from 2 A-frags + 2 B-frags. 8 waves/block in a 2x4
// arrangement -> 64x128 block tile.
template <bool HALF_OUT>
__global__ __launch_bounds__(256) void k_gemm_wmma(
    const _Float16* __restrict__ A,   // row-major (Mr x K)
    const _Float16* __restrict__ Bt,  // row-major (N x K)  == B transposed
    const float*    __restrict__ bias,
    void*           __restrict__ Cout,
    int Mr, int N, int K) {
    const int lane = threadIdx.x & 31;
    const int wave = threadIdx.x >> 5;
    const int wr   = wave >> 2;   // 0..1
    const int wc   = wave & 3;    // 0..3
    const int lh   = lane >> 4;   // half-group 0/1
    const int ln   = lane & 15;

    const int r0 = blockIdx.x * 64 + wr * 32;   // wave row base
    const int c0 = blockIdx.y * 128 + wc * 32;  // wave col base

    const _Float16* arow0 = A  + (size_t)(r0 + ln) * K;
    const _Float16* arow1 = A  + (size_t)(r0 + 16 + ln) * K;
    const _Float16* brow0 = Bt + (size_t)(c0 + ln) * K;
    const _Float16* brow1 = Bt + (size_t)(c0 + 16 + ln) * K;

    v8f acc00 = {}, acc01 = {}, acc10 = {}, acc11 = {};
    #pragma unroll 2
    for (int k0 = 0; k0 < K; k0 += 32) {
        Frag16 a0, a1, b0, b1;
        // A-operand layout: halves 0-7 = K[k0+8*lh .. +7], halves 8-15 = +16
        a0.h8[0] = *(const v8h*)(arow0 + k0 + lh * 8);
        a0.h8[1] = *(const v8h*)(arow0 + k0 + 16 + lh * 8);
        a1.h8[0] = *(const v8h*)(arow1 + k0 + lh * 8);
        a1.h8[1] = *(const v8h*)(arow1 + k0 + 16 + lh * 8);
        // B-operand layout: halves 0-7 = K[k0+8*lh .. +7], halves 8-15 = +16
        b0.h8[0] = *(const v8h*)(brow0 + k0 + lh * 8);
        b0.h8[1] = *(const v8h*)(brow0 + k0 + 16 + lh * 8);
        b1.h8[0] = *(const v8h*)(brow1 + k0 + lh * 8);
        b1.h8[1] = *(const v8h*)(brow1 + k0 + 16 + lh * 8);
        acc00 = wmma_f16(a0.h16, b0.h16, acc00);
        acc01 = wmma_f16(a0.h16, b1.h16, acc01);
        acc10 = wmma_f16(a1.h16, b0.h16, acc10);
        acc11 = wmma_f16(a1.h16, b1.h16, acc11);
    }

    const float bn0 = bias ? bias[c0 + ln] : 0.0f;
    const float bn1 = bias ? bias[c0 + 16 + ln] : 0.0f;
    #pragma unroll
    for (int r = 0; r < 8; ++r) {
        int rowA = r0 + lh * 8 + r;       // C layout: lanes0-15 rows0-7, lanes16-31 rows8-15
        int rowB = r0 + 16 + lh * 8 + r;
        float v00 = acc00[r] + bn0, v01 = acc01[r] + bn1;
        float v10 = acc10[r] + bn0, v11 = acc11[r] + bn1;
        if (HALF_OUT) {
            _Float16* C = (_Float16*)Cout;
            C[(size_t)rowA * N + c0 + ln]      = (_Float16)v00;
            C[(size_t)rowA * N + c0 + 16 + ln] = (_Float16)v01;
            C[(size_t)rowB * N + c0 + ln]      = (_Float16)v10;
            C[(size_t)rowB * N + c0 + 16 + ln] = (_Float16)v11;
        } else {
            float* C = (float*)Cout;
            C[(size_t)rowA * N + c0 + ln]      = v00;
            C[(size_t)rowA * N + c0 + 16 + ln] = v01;
            C[(size_t)rowB * N + c0 + ln]      = v10;
            C[(size_t)rowB * N + c0 + 16 + ln] = v11;
        }
    }
}

// ---------------- RoPE + pack:  raw (B*M x D) f16 ->  Q/K (B,H,M,Hd), Vt (B,H,Hd,M)
__global__ __launch_bounds__(256) void k_rope_pack(
    const _Float16* __restrict__ Qr, const _Float16* __restrict__ Kr,
    const _Float16* __restrict__ Vr,
    const float* __restrict__ cy, const float* __restrict__ sy,
    const float* __restrict__ cx, const float* __restrict__ sx,
    _Float16* __restrict__ Qh, _Float16* __restrict__ Kh,
    _Float16* __restrict__ Vt) {
    size_t idx = (size_t)blockIdx.x * 256 + threadIdx.x;
    if (idx >= NMD) return;
    int    col = (int)(idx % DD);
    size_t row = idx / DD;
    int m = (int)(row % MM);
    int b = (int)(row / MM);
    int h = col / HD;
    int d = col % HD;

    float q  = (float)Qr[idx];
    float k  = (float)Kr[idx];
    float qn = (float)Qr[idx ^ 1];  // rotate_half partner (d^1, same parity trick)
    float kn = (float)Kr[idx ^ 1];
    float v  = (float)Vr[idx];

    int dd   = d & (HALFD - 1);
    bool isx = d >= HALFD;
    float c = (isx ? cx : cy)[(size_t)m * HALFD + dd];
    float s = (isx ? sx : sy)[(size_t)m * HALFD + dd];
    float sgn = (d & 1) ? 1.0f : -1.0f;

    float q2 = q * c + sgn * qn * s;
    float k2 = k * c + sgn * kn * s;

    size_t bh = (size_t)b * HH + h;
    size_t qk = (bh * MM + m) * HD + d;
    Qh[qk] = (_Float16)(q2 * 0.125f);  // fold softmax scale (Hd^-0.5) into Q
    Kh[qk] = (_Float16)k2;
    Vt[(bh * HD + d) * MM + m] = (_Float16)v;  // transposed for B-operand loads
}

// ---------------- flash attention: one 32-query tile per wave -------------
// Per 32-key chunk: 8 score wmma + 8 PV wmma reusing each K/V fragment across
// two query row tiles (~32 FLOP per L2 byte). P staged via a per-wave LDS
// slab to convert C-layout -> A-layout, guarded by s_wait_dscnt.
__global__ __launch_bounds__(256) void k_flash_attn(
    const _Float16* __restrict__ Qh,  // (B,H,M,Hd)
    const _Float16* __restrict__ Kh,  // (B,H,M,Hd)
    const _Float16* __restrict__ Vt,  // (B,H,Hd,M)
    _Float16* __restrict__ Aout) {    // (B,M,D) f16
    __shared__ _Float16 plds[8][32 * 32];

    const int lane = threadIdx.x & 31;
    const int wave = threadIdx.x >> 5;
    const int bh   = blockIdx.x >> 3;               // 0..191
    const int qt   = (blockIdx.x & 7) * 8 + wave;   // 0..63 (32-row tiles)
    const int b    = bh / HH;
    const int h    = bh % HH;
    const int lh   = lane >> 4;
    const int ln   = lane & 15;

    const _Float16* Qb = Qh + (size_t)bh * MM * HD;
    const _Float16* Kb = Kh + (size_t)bh * MM * HD;
    const _Float16* Vb = Vt + (size_t)bh * HD * MM;

    // Q fragments: 2 row tiles x 2 K-steps (Hd = 64), live for the whole loop
    Frag16 qf[2][2];
    #pragma unroll
    for (int rt = 0; rt < 2; ++rt) {
        const _Float16* qrow = Qb + (size_t)(qt * 32 + rt * 16 + ln) * HD;
        qf[rt][0].h8[0] = *(const v8h*)(qrow + lh * 8);
        qf[rt][0].h8[1] = *(const v8h*)(qrow + 16 + lh * 8);
        qf[rt][1].h8[0] = *(const v8h*)(qrow + 32 + lh * 8);
        qf[rt][1].h8[1] = *(const v8h*)(qrow + 48 + lh * 8);
    }

    v8f o[2][4];
    float mrow[2][8], lrow[2][8];
    #pragma unroll
    for (int rt = 0; rt < 2; ++rt) {
        #pragma unroll
        for (int t = 0; t < 4; ++t) o[rt][t] = (v8f){};
        #pragma unroll
        for (int r = 0; r < 8; ++r) { mrow[rt][r] = -1e30f; lrow[rt][r] = 0.0f; }
    }

    _Float16* pl = plds[wave];

    for (int j0 = 0; j0 < MM; j0 += 32) {
        // ---- K fragments for this chunk: 2 col tiles x 2 K-steps
        Frag16 kf[2][2];
        #pragma unroll
        for (int ct = 0; ct < 2; ++ct) {
            const _Float16* krow = Kb + (size_t)(j0 + ct * 16 + ln) * HD;
            kf[ct][0].h8[0] = *(const v8h*)(krow + lh * 8);
            kf[ct][0].h8[1] = *(const v8h*)(krow + 16 + lh * 8);
            kf[ct][1].h8[0] = *(const v8h*)(krow + 32 + lh * 8);
            kf[ct][1].h8[1] = *(const v8h*)(krow + 48 + lh * 8);
        }

        // ---- scores: 4 tiles of 16x16, each accumulated over Hd in 2 steps
        v8f s[2][2];
        #pragma unroll
        for (int rt = 0; rt < 2; ++rt)
            #pragma unroll
            for (int ct = 0; ct < 2; ++ct) {
                v8f acc = {};
                acc = wmma_f16(qf[rt][0].h16, kf[ct][0].h16, acc);
                acc = wmma_f16(qf[rt][1].h16, kf[ct][1].h16, acc);
                s[rt][ct] = acc;
            }

        // ---- online softmax over this 32-key chunk (16 rows per row tile)
        #pragma unroll
        for (int rt = 0; rt < 2; ++rt) {
            #pragma unroll
            for (int r = 0; r < 8; ++r) {
                float v0 = s[rt][0][r], v1 = s[rt][1][r];
                float cm = redmax16(fmaxf(v0, v1));
                float nm = fmaxf(mrow[rt][r], cm);
                float al = __expf(mrow[rt][r] - nm);
                float p0 = __expf(v0 - nm);
                float p1 = __expf(v1 - nm);
                float ps = redsum16(p0 + p1);
                lrow[rt][r] = lrow[rt][r] * al + ps;
                mrow[rt][r] = nm;
                o[rt][0][r] *= al; o[rt][1][r] *= al;
                o[rt][2][r] *= al; o[rt][3][r] *= al;
                int prow = rt * 16 + lh * 8 + r;
                pl[prow * 32 + ln]      = (_Float16)p0;
                pl[prow * 32 + 16 + ln] = (_Float16)p1;
            }
        }
        // wave-private LDS slab: drain this wave's DS ops before reading back
        asm volatile("s_wait_dscnt 0x0" ::: "memory");

        // ---- read P back in A-operand layout (two 16x32 tiles, K = 32 keys)
        Frag16 pf[2];
        #pragma unroll
        for (int rt = 0; rt < 2; ++rt) {
            const _Float16* prow = pl + (rt * 16 + ln) * 32;
            pf[rt].h8[0] = *(const v8h*)(prow + lh * 8);
            pf[rt].h8[1] = *(const v8h*)(prow + 16 + lh * 8);
        }

        // ---- O += P * V : each V fragment reused across both row tiles
        #pragma unroll
        for (int t = 0; t < 4; ++t) {
            Frag16 vf;
            const _Float16* vrow = Vb + (size_t)(t * 16 + ln) * MM + j0;
            vf.h8[0] = *(const v8h*)(vrow + lh * 8);
            vf.h8[1] = *(const v8h*)(vrow + 16 + lh * 8);
            o[0][t] = wmma_f16(pf[0].h16, vf.h16, o[0][t]);
            o[1][t] = wmma_f16(pf[1].h16, vf.h16, o[1][t]);
        }
    }

    // ---- normalize and store to (B, M, H*Hd) f16 for the output projection
    #pragma unroll
    for (int rt = 0; rt < 2; ++rt) {
        size_t obase = ((size_t)b * MM + qt * 32 + rt * 16 + lh * 8) * DD +
                       (size_t)h * HD;
        #pragma unroll
        for (int r = 0; r < 8; ++r) {
            float inv = 1.0f / lrow[rt][r];
            size_t ob = obase + (size_t)r * DD;
            Aout[ob + 0 * 16 + ln] = (_Float16)(o[rt][0][r] * inv);
            Aout[ob + 1 * 16 + ln] = (_Float16)(o[rt][1][r] * inv);
            Aout[ob + 2 * 16 + ln] = (_Float16)(o[rt][2][r] * inv);
            Aout[ob + 3 * 16 + ln] = (_Float16)(o[rt][3][r] * inv);
        }
    }
}

// ---------------- host-side launcher ----------------
extern "C" void kernel_launch(void* const* d_in, const int* in_sizes, int n_in,
                              void* d_out, int out_size, void* d_ws, size_t ws_size,
                              hipStream_t stream) {
    const float* x  = (const float*)d_in[0];
    const float* cy = (const float*)d_in[1];
    const float* sy = (const float*)d_in[2];
    const float* cx = (const float*)d_in[3];
    const float* sx = (const float*)d_in[4];
    const float* Wq = (const float*)d_in[5];
    const float* bq = (const float*)d_in[6];
    const float* Wk = (const float*)d_in[7];
    const float* bk = (const float*)d_in[8];
    const float* Wv = (const float*)d_in[9];
    const float* bv = (const float*)d_in[10];
    const float* Wo = (const float*)d_in[11];
    const float* bo = (const float*)d_in[12];

    const size_t WSZ = (size_t)DD * DD;  // 589824 elements per weight matrix

    _Float16* ws  = (_Float16*)d_ws;
    _Float16* xh  = ws;                 // (NM x D) f16
    _Float16* WqT = xh  + NMD;          // (N x K) f16 transposed weights
    _Float16* WkT = WqT + WSZ;
    _Float16* WvT = WkT + WSZ;
    _Float16* WoT = WvT + WSZ;
    _Float16* Qraw = WoT + WSZ;         // raw projection outputs f16
    _Float16* Kraw = Qraw + NMD;
    _Float16* Vraw = Kraw + NMD;
    _Float16* Qh   = Vraw + NMD;        // packed (B,H,M,Hd)
    _Float16* Kh   = Qh + NMD;
    _Float16* Vt   = Kh + NMD;          // (B,H,Hd,M)
    _Float16* attn = xh;                // alias: xh is dead after projections

    // 1) x -> f16
    k_cvt_f32_to_f16<<<(unsigned)(NMD / 256), 256, 0, stream>>>(x, xh, NMD);

    // 2) transpose + cast weights
    {
        unsigned g = (unsigned)((WSZ + 255) / 256);
        k_w_transpose_f16<<<g, 256, 0, stream>>>(Wq, WqT, DD);
        k_w_transpose_f16<<<g, 256, 0, stream>>>(Wk, WkT, DD);
        k_w_transpose_f16<<<g, 256, 0, stream>>>(Wv, WvT, DD);
        k_w_transpose_f16<<<g, 256, 0, stream>>>(Wo, WoT, DD);
    }

    // 3) Q/K/V projections (WMMA f16 -> f32 accum -> f16 out)
    dim3 ggrid(NM / 64, DD / 128);  // (512, 6), 64x128 block tiles
    k_gemm_wmma<true><<<ggrid, 256, 0, stream>>>(xh, WqT, bq, Qraw, NM, DD, DD);
    k_gemm_wmma<true><<<ggrid, 256, 0, stream>>>(xh, WkT, bk, Kraw, NM, DD, DD);
    k_gemm_wmma<true><<<ggrid, 256, 0, stream>>>(xh, WvT, bv, Vraw, NM, DD, DD);

    // 4) RoPE + head packing + V transpose
    k_rope_pack<<<(unsigned)(NMD / 256), 256, 0, stream>>>(
        Qraw, Kraw, Vraw, cy, sy, cx, sx, Qh, Kh, Vt);

    // 5) flash attention: 192 head-batches x 64 q-tiles (32 rows), 8 waves/block
    k_flash_attn<<<BB * HH * 8, 256, 0, stream>>>(Qh, Kh, Vt, attn);

    // 6) output projection -> f32 d_out
    k_gemm_wmma<false><<<ggrid, 256, 0, stream>>>(attn, WoT, bo, (float*)d_out,
                                                  NM, DD, DD);
}
